// MultiHeadAttention_83880711291014
// MI455X (gfx1250) — matrix operand order repaired and checked by reading, not compiled, not run on hardware
//
#include <hip/hip_runtime.h>
#include <hip/hip_bf16.h>
#include <math.h>

#define B_  2
#define S_  2048
#define D_  1024
#define H_  16
#define DK_ 64
#define DV_ 64

typedef __bf16 bfrag4  __attribute__((ext_vector_type(4)));
typedef __bf16 bfrag8  __attribute__((ext_vector_type(8)));
typedef __bf16 bfrag16 __attribute__((ext_vector_type(16)));
typedef float  ffrag4  __attribute__((ext_vector_type(4)));
typedef float  ffrag8  __attribute__((ext_vector_type(8)));

// ---------------------------------------------------------------------------
// Fragment loaders per CDNA5 WMMA VGPR layouts (CDNA5 ISA §7.12.2)
// A (16x32 bf16): lanes 0-15 row M=lane hold K=0..7 then K=16..23;
//                 lanes 16-31 hold K=8..15 then K=24..31  -> two 16B loads.
// B (32x16 bf16): lane%16 = column N, same K order as A (K=8h..8h+7 then
//                 16+8h..) -> two 16B loads when B is stored column-major.
// C/D (16x16 f32): VGPR r = row (lane<16 ? r : r+8), col = lane%16.
// ---------------------------------------------------------------------------

__device__ __forceinline__ bfrag16 load_a_frag(const __bf16* base, int lda) {
  const int lane = threadIdx.x & 31;
  const int hl = lane >> 4, lo = lane & 15;
  const __bf16* p = base + (size_t)lo * lda + hl * 8;
  bfrag8 x0 = *(const bfrag8*)(p);
  bfrag8 x1 = *(const bfrag8*)(p + 16);
  bfrag16 r;
#pragma unroll
  for (int i = 0; i < 8; ++i) { r[i] = x0[i]; r[i + 8] = x1[i]; }
  return r;
}

__device__ __forceinline__ bfrag16 load_b_frag(const __bf16* base, int ldb) {
  const int lane = threadIdx.x & 31;
  const int hl = lane >> 4, lo = lane & 15;
  const __bf16* p = base + (size_t)lo * ldb + hl * 8;
  bfrag8 x0 = *(const bfrag8*)(p);
  bfrag8 x1 = *(const bfrag8*)(p + 16);
  bfrag16 r;
#pragma unroll
  for (int i = 0; i < 8; ++i) { r[i] = x0[i]; r[i + 8] = x1[i]; }
  return r;
}

__device__ __forceinline__ ffrag8 wmma_bf16(bfrag16 a, bfrag16 b, ffrag8 c) {
  return __builtin_amdgcn_wmma_f32_16x16x32_bf16(false, a, false, b, (short)0, c,
                                                 false, false);
}

// ---------------------------------------------------------------------------
// Precision staging (fp32 -> bf16, weight transposes for K-contiguous B frags)
// ---------------------------------------------------------------------------

__global__ void cvt_kernel(const float* __restrict__ in, __bf16* __restrict__ out,
                           int n4) {  // n4 = n/4
  int i = blockIdx.x * blockDim.x + threadIdx.x;
  if (i >= n4) return;
  ffrag4 x = ((const ffrag4*)in)[i];
  bfrag4 y;
#pragma unroll
  for (int j = 0; j < 4; ++j) y[j] = (__bf16)x[j];
  ((bfrag4*)out)[i] = y;
}

// in: [H, D, DK] -> out: [H, DK, D]
__global__ void trw_kernel(const float* __restrict__ in, __bf16* __restrict__ out) {
  int i = blockIdx.x * blockDim.x + threadIdx.x;
  if (i >= H_ * D_ * DK_) return;
  int k = i % DK_;
  int d = (i / DK_) % D_;
  int h = i / (DK_ * D_);
  out[((size_t)h * DK_ + k) * D_ + d] = (__bf16)in[i];
}

// in: [H*DV, D] -> out: [D, H*DV]
__global__ void trwo_kernel(const float* __restrict__ in, __bf16* __restrict__ out) {
  int i = blockIdx.x * blockDim.x + threadIdx.x;
  if (i >= H_ * DV_ * D_) return;
  int d  = i % D_;
  int hv = i / D_;
  out[(size_t)d * (H_ * DV_) + hv] = (__bf16)in[i];
}

// ---------------------------------------------------------------------------
// Generic double-buffered 16x64 strip GEMM core: A row-major [lda=K], B
// stored column-major ([64 rows of Kdim]), fp32 accumulate. Distinct "cur"
// and "nxt" fragment registers let the scheduler overlap next-step loads with
// current-step WMMAs (partial s_wait_loadcnt instead of full drains).
// ---------------------------------------------------------------------------

__device__ __forceinline__ void
gemm_strip_16x64(const __bf16* __restrict__ Abase, int lda,
                 const __bf16* __restrict__ Bbase, int ldb, int Kdim,
                 ffrag8 acc[4]) {
  bfrag16 a_cur = load_a_frag(Abase, lda);
  bfrag16 b_cur0 = load_b_frag(Bbase + (size_t)0  * ldb, ldb);
  bfrag16 b_cur1 = load_b_frag(Bbase + (size_t)16 * ldb, ldb);
  bfrag16 b_cur2 = load_b_frag(Bbase + (size_t)32 * ldb, ldb);
  bfrag16 b_cur3 = load_b_frag(Bbase + (size_t)48 * ldb, ldb);
  for (int d = 0; d < Kdim - 32; d += 32) {
    bfrag16 a_nxt  = load_a_frag(Abase + d + 32, lda);
    bfrag16 b_nxt0 = load_b_frag(Bbase + (size_t)0  * ldb + d + 32, ldb);
    bfrag16 b_nxt1 = load_b_frag(Bbase + (size_t)16 * ldb + d + 32, ldb);
    bfrag16 b_nxt2 = load_b_frag(Bbase + (size_t)32 * ldb + d + 32, ldb);
    bfrag16 b_nxt3 = load_b_frag(Bbase + (size_t)48 * ldb + d + 32, ldb);
    acc[0] = wmma_bf16(a_cur, b_cur0, acc[0]);
    acc[1] = wmma_bf16(a_cur, b_cur1, acc[1]);
    acc[2] = wmma_bf16(a_cur, b_cur2, acc[2]);
    acc[3] = wmma_bf16(a_cur, b_cur3, acc[3]);
    a_cur = a_nxt;
    b_cur0 = b_nxt0; b_cur1 = b_nxt1; b_cur2 = b_nxt2; b_cur3 = b_nxt3;
  }
  acc[0] = wmma_bf16(a_cur, b_cur0, acc[0]);
  acc[1] = wmma_bf16(a_cur, b_cur1, acc[1]);
  acc[2] = wmma_bf16(a_cur, b_cur2, acc[2]);
  acc[3] = wmma_bf16(a_cur, b_cur3, acc[3]);
}

// ---------------------------------------------------------------------------
// QKV projection: one wave = one 16x64 strip (all DK cols of one head for a
// 16-row block). transpose==0 : Out[b,h,s,n]   transpose==1 : Out[b,h,n,s]
// ---------------------------------------------------------------------------

__global__ void __launch_bounds__(256)
proj_kernel(const __bf16* __restrict__ X,    // [B,S,D] bf16
            const __bf16* __restrict__ Wt,   // [H,64,D] bf16 (K-contiguous)
            const float*  __restrict__ bias, // [H,64] fp32
            __bf16* __restrict__ Out, int transpose) {
  const int lane = threadIdx.x & 31;
  const int hl = lane >> 4, lo = lane & 15;
  int wid = blockIdx.x * 8 + (threadIdx.x >> 5);     // 4096 waves total
  const int sblk = wid & 127; wid >>= 7;
  const int h    = wid & 15;  wid >>= 4;
  const int b    = wid;
  const int s0 = sblk * 16;

  const __bf16* Abase = X  + ((size_t)b * S_ + s0) * D_;
  const __bf16* Bbase = Wt + (size_t)h * DK_ * D_;

  ffrag8 acc[4];
  {
    ffrag8 z = {};
#pragma unroll
    for (int t = 0; t < 4; ++t) acc[t] = z;
  }
  gemm_strip_16x64(Abase, D_, Bbase, D_, D_, acc);

  const size_t bh = (size_t)b * H_ + h;
#pragma unroll
  for (int t = 0; t < 4; ++t) {
    const int n = t * 16 + lo;
    const float bv = bias[h * DK_ + n];
#pragma unroll
    for (int r = 0; r < 8; ++r) {
      float val = acc[t][r] + bv;
      int row = s0 + hl * 8 + r;
      if (!transpose)
        Out[(bh * S_ + row) * DK_ + n] = (__bf16)val;
      else
        Out[(bh * DV_ + n) * S_ + row] = (__bf16)val;
    }
  }
}

// ---------------------------------------------------------------------------
// Flash attention: one wave owns 16 query rows of one (b,h); streams keys in
// blocks of 32 with online softmax. Never materializes [S,S] scores.
// ---------------------------------------------------------------------------

__global__ void __launch_bounds__(256)
attn_kernel(const __bf16* __restrict__ Qp,   // [B,H,S,64]
            const __bf16* __restrict__ Kp,   // [B,H,S,64]
            const __bf16* __restrict__ Vt,   // [B,H,64,S]
            const float*  __restrict__ mask, // [S,S]
            __bf16* __restrict__ Abuf) {     // [B,S,H*64] concat-head output
  __shared__ __bf16 smemP[8][16 * 32];       // per-wave P tile (16x32), 8 KB

  const int lane = threadIdx.x & 31;
  const int hl = lane >> 4, lo = lane & 15;
  const int wave = threadIdx.x >> 5;
  const int wid  = blockIdx.x * 8 + wave;
  const int qblk = wid % (S_ / 16);
  const int bh   = wid / (S_ / 16);
  const int h = bh % H_, b = bh / H_;
  const int q0 = qblk * 16;

  const __bf16* Qbase = Qp + ((size_t)bh * S_ + q0) * DK_;
  const __bf16* Kbase = Kp + (size_t)bh * S_ * DK_;
  const __bf16* Vbase = Vt + (size_t)bh * DV_ * S_;

  // Q fragments loaded once (16 x 64 = two A fragments)
  const bfrag16 qf0 = load_a_frag(Qbase + 0, DK_);
  const bfrag16 qf1 = load_a_frag(Qbase + 32, DK_);

  float m_row[8], l_row[8];
  ffrag8 acc[4];
  {
    ffrag8 z = {};
#pragma unroll
    for (int t = 0; t < 4; ++t) acc[t] = z;
#pragma unroll
    for (int r = 0; r < 8; ++r) { m_row[r] = -INFINITY; l_row[r] = 0.f; }
  }

  __bf16* Pl = &smemP[wave][0];

  for (int t0 = 0; t0 < S_; t0 += 32) {
    if (t0 + 32 < S_) {                       // gfx1250 global_prefetch_b8
      __builtin_prefetch(Kbase + (size_t)(t0 + 32) * DK_, 0, 1);
      __builtin_prefetch(Vbase + t0 + 32, 0, 1);
    }
    // ---- scores: two 16x16 tiles over 32 keys, K-loop over DK=64 ----
    ffrag8 s0 = {}, s1 = {};
    {
      bfrag16 k00 = load_b_frag(Kbase + (size_t)t0 * DK_ + 0, DK_);
      bfrag16 k01 = load_b_frag(Kbase + (size_t)t0 * DK_ + 32, DK_);
      bfrag16 k10 = load_b_frag(Kbase + (size_t)(t0 + 16) * DK_ + 0, DK_);
      bfrag16 k11 = load_b_frag(Kbase + (size_t)(t0 + 16) * DK_ + 32, DK_);
      s0 = wmma_bf16(qf0, k00, s0);
      s0 = wmma_bf16(qf1, k01, s0);
      s1 = wmma_bf16(qf0, k10, s1);
      s1 = wmma_bf16(qf1, k11, s1);
    }
    // ---- scale + additive mask ----
    float p0[8], p1[8];
#pragma unroll
    for (int r = 0; r < 8; ++r) {
      const int row = q0 + hl * 8 + r;
      p0[r] = s0[r] * 0.125f + mask[(size_t)row * S_ + t0 + lo];
      p1[r] = s1[r] * 0.125f + mask[(size_t)row * S_ + t0 + 16 + lo];
    }
    // ---- online softmax (row spans 16 lanes of a half; xor<16 stays inside) --
#pragma unroll
    for (int r = 0; r < 8; ++r) {
      float mb = fmaxf(p0[r], p1[r]);
#pragma unroll
      for (int off = 1; off < 16; off <<= 1)
        mb = fmaxf(mb, __shfl_xor(mb, off, 32));
      const float mn    = fmaxf(m_row[r], mb);
      const float alpha = __expf(m_row[r] - mn);
      m_row[r] = mn;
      p0[r] = __expf(p0[r] - mn);
      p1[r] = __expf(p1[r] - mn);
      float bs = p0[r] + p1[r];
#pragma unroll
      for (int off = 1; off < 16; off <<= 1)
        bs += __shfl_xor(bs, off, 32);
      l_row[r] = l_row[r] * alpha + bs;
#pragma unroll
      for (int t = 0; t < 4; ++t) acc[t][r] *= alpha;
    }
    // ---- C-layout -> A-layout via LDS (cross-lane), then P @ V ----
#pragma unroll
    for (int r = 0; r < 8; ++r) {
      Pl[(hl * 8 + r) * 32 + lo]      = (__bf16)p0[r];
      Pl[(hl * 8 + r) * 32 + lo + 16] = (__bf16)p1[r];
    }
    asm volatile("s_wait_dscnt 0x0" ::: "memory");  // cross-lane LDS hazard
    const bfrag16 pf = load_a_frag(Pl, 32);
#pragma unroll
    for (int t = 0; t < 4; ++t) {
      bfrag16 vf = load_b_frag(Vbase + (size_t)(t * 16) * S_ + t0, S_);
      acc[t] = wmma_bf16(pf, vf, acc[t]);
    }
  }
  // ---- epilogue: normalize, store concat-head bf16 row ----
#pragma unroll
  for (int t = 0; t < 4; ++t) {
#pragma unroll
    for (int r = 0; r < 8; ++r) {
      const int row = q0 + hl * 8 + r;
      const int col = t * 16 + lo;
      const float o = acc[t][r] / l_row[r];
      Abuf[((size_t)b * S_ + row) * (H_ * DV_) + h * DV_ + col] = (__bf16)o;
    }
  }
}

// ---------------------------------------------------------------------------
// Output projection: one wave = 16x64 strip of [B*S, D], double-buffered.
// [B*S, H*DV] x [H*DV, D] + bo -> fp32 d_out
// ---------------------------------------------------------------------------

__global__ void __launch_bounds__(256)
out_gemm_kernel(const __bf16* __restrict__ A,    // [B*S, 1024]
                const __bf16* __restrict__ Wt,   // [1024(D), 1024(HDV)]
                const float*  __restrict__ bias, // [D]
                float* __restrict__ Out) {       // [B*S, D]
  const int lane = threadIdx.x & 31;
  const int hl = lane >> 4, lo = lane & 15;
  int wid = blockIdx.x * 8 + (threadIdx.x >> 5);  // 4096 waves total
  const int nquad = wid & 15; wid >>= 4;          // 16 quads of 64 cols
  const int m0 = wid * 16, n0 = nquad * 64;
  const int K = H_ * DV_;

  ffrag8 acc[4];
  {
    ffrag8 z = {};
#pragma unroll
    for (int t = 0; t < 4; ++t) acc[t] = z;
  }
  gemm_strip_16x64(A + (size_t)m0 * K, K, Wt + (size_t)n0 * K, K, K, acc);

#pragma unroll
  for (int t = 0; t < 4; ++t) {
    const int n = n0 + t * 16 + lo;
    const float bv = bias[n];
#pragma unroll
    for (int r = 0; r < 8; ++r)
      Out[((size_t)m0 + hl * 8 + r) * D_ + n] = acc[t][r] + bv;
  }
}

// ---------------------------------------------------------------------------

extern "C" void kernel_launch(void* const* d_in, const int* in_sizes, int n_in,
                              void* d_out, int out_size, void* d_ws, size_t ws_size,
                              hipStream_t stream) {
  (void)in_sizes; (void)n_in; (void)out_size; (void)ws_size;
  const float* q    = (const float*)d_in[0];
  const float* k    = (const float*)d_in[1];
  const float* v    = (const float*)d_in[2];
  const float* mask = (const float*)d_in[3];
  const float* Wq   = (const float*)d_in[4];
  const float* bq   = (const float*)d_in[5];
  const float* Wk   = (const float*)d_in[6];
  const float* bk   = (const float*)d_in[7];
  const float* Wv   = (const float*)d_in[8];
  const float* bv   = (const float*)d_in[9];
  const float* Wo   = (const float*)d_in[10];
  const float* bo   = (const float*)d_in[11];
  float* Out = (float*)d_out;

  char* ws = (char*)d_ws;
  size_t off = 0;
  auto alloc = [&](size_t bytes) {
    void* p = ws + off;
    off += (bytes + 255) & ~(size_t)255;
    return p;
  };
  const size_t NX = (size_t)B_ * S_ * D_;          // 4,194,304
  const size_t NW = (size_t)H_ * D_ * DK_;         // 1,048,576
  const size_t NP = (size_t)B_ * H_ * S_ * DK_;    // 4,194,304

  __bf16* Xq   = (__bf16*)alloc(NX * 2);
  __bf16* Xk   = (__bf16*)alloc(NX * 2);
  __bf16* Xv   = (__bf16*)alloc(NX * 2);
  __bf16* Wqt  = (__bf16*)alloc(NW * 2);
  __bf16* Wkt  = (__bf16*)alloc(NW * 2);
  __bf16* Wvt  = (__bf16*)alloc(NW * 2);
  __bf16* Wot  = (__bf16*)alloc(NW * 2);
  __bf16* Qp   = (__bf16*)alloc(NP * 2);
  __bf16* Kp   = (__bf16*)alloc(NP * 2);
  __bf16* Vt   = (__bf16*)alloc(NP * 2);
  __bf16* Abuf = (__bf16*)alloc(NP * 2);

  const int TB = 256;
  // stage activations + weights to bf16 (one-time, halves all GEMM traffic)
  cvt_kernel<<<(int)(NX / 4 / TB), TB, 0, stream>>>(q, Xq, (int)(NX / 4));
  cvt_kernel<<<(int)(NX / 4 / TB), TB, 0, stream>>>(k, Xk, (int)(NX / 4));
  cvt_kernel<<<(int)(NX / 4 / TB), TB, 0, stream>>>(v, Xv, (int)(NX / 4));
  trw_kernel<<<(int)(NW / TB), TB, 0, stream>>>(Wq, Wqt);
  trw_kernel<<<(int)(NW / TB), TB, 0, stream>>>(Wk, Wkt);
  trw_kernel<<<(int)(NW / TB), TB, 0, stream>>>(Wv, Wvt);
  trwo_kernel<<<(int)(NW / TB), TB, 0, stream>>>(Wo, Wot);

  // per-head projections: 4096 wave-strips (16 rows x 64 cols) / 8 per block
  proj_kernel<<<512, TB, 0, stream>>>(Xq, Wqt, bq, Qp, 0);
  proj_kernel<<<512, TB, 0, stream>>>(Xk, Wkt, bk, Kp, 0);
  proj_kernel<<<512, TB, 0, stream>>>(Xv, Wvt, bv, Vt, 1);  // V stored [b,h,v,t]

  // flash attention: 4096 q-row-block waves / 8 per block
  attn_kernel<<<512, TB, 0, stream>>>(Qp, Kp, Vt, mask, Abuf);

  // output projection -> fp32 (4096 wave-strips / 8 per block)
  out_gemm_kernel<<<512, TB, 0, stream>>>(Abuf, Wot, bo, Out);
}
